// MoCTopKExperts_78615081386224
// MI455X (gfx1250) — compile-verified, matrix operand order checked by reading, not run on hardware
//
#include <hip/hip_runtime.h>
#include <hip/hip_bf16.h>
#include <math.h>

// ---------------- constants from the reference problem ----------------
#define DD      768       // model dim
#define EE      8         // experts
#define KK      2         // top-k
#define HFF     2048      // expert hidden
#define NHEADS  4
#define DHD     192       // head dim
#define TD      2304      // 3*D
#define CHUNK   2048      // tokens per collab chunk
#define LDS_CHUNK 4096    // k_rank LDS tile (32KB)

// ---------------- WMMA types / helpers (gfx1250, wave32) ----------------
typedef __attribute__((ext_vector_type(16))) _Float16 v16h;
typedef __attribute__((ext_vector_type(8)))  _Float16 v8h;
typedef __attribute__((ext_vector_type(4)))  _Float16 v4h;
typedef __attribute__((ext_vector_type(8)))  float    v8f;

__device__ __forceinline__ v8f wmma_f16(v16h a, v16h b, v8f c) {
  // D = A(16x32 f16) * B(32x16 f16) + C(16x16 f32)
  return __builtin_amdgcn_wmma_f32_16x16x32_f16(false, a, false, b, (short)0, c,
                                                false, false);
}

// 16-bit A-matrix 16x32 lane layout (ISA 7.12.2):
// lane half = lane>>4 adds K+8; VGPR v: grp=(v>>2)*16, pair=(v&3)*2.
// => f[0:8]  = row[k0 .. k0+7], f[8:16] = row[k0+16 .. k0+23]  (k0 = kt+half*8)
// Same layout used for B columns (B row-major N x K => A @ B^T).
__device__ __forceinline__ v16h load_fragA(const float* row, int k0) {
  v16h f;
#pragma unroll
  for (int v = 0; v < 8; ++v) {
    int k = k0 + ((v >> 2) << 4) + ((v & 3) << 1);
    float2 t = *(const float2*)(row + k);
    f[2 * v]     = (_Float16)t.x;
    f[2 * v + 1] = (_Float16)t.y;
  }
  return f;
}

__device__ __forceinline__ v16h load_fragB(const _Float16* row, int k0) {
  v8h lo = *(const v8h*)(row + k0);        // K = k0 .. k0+7
  v8h hi = *(const v8h*)(row + k0 + 16);   // K = k0+16 .. k0+23
  return __builtin_shufflevector(lo, hi, 0, 1, 2, 3, 4, 5, 6, 7,
                                 8, 9, 10, 11, 12, 13, 14, 15);
}

__device__ __forceinline__ float wave_sum(float v) {
#pragma unroll
  for (int o = 16; o > 0; o >>= 1) v += __shfl_xor(v, o, 32);
  return v;
}

// ---------------- f32 -> f16 weight pre-conversion (once per call) ------------
__global__ __launch_bounds__(256)
void k_cvt16(const float* src, _Float16* dst) {
  size_t i = (size_t)blockIdx.x * blockDim.x + threadIdx.x;
  float4 v = ((const float4*)src)[i];
  v4h h;
  h[0] = (_Float16)v.x; h[1] = (_Float16)v.y;
  h[2] = (_Float16)v.z; h[3] = (_Float16)v.w;
  ((v4h*)dst)[i] = h;
}

// ---------------- gating + aux accumulation ----------------
__global__ __launch_bounds__(256)
void k_gate(const float* x, const float* gate_w, int* te, float* prio,
            float* tpq, float* accum, int Ntok) {
  __shared__ float sAcc[32];
  int t = threadIdx.x;
  if (t < 32) sAcc[t] = 0.f;
  __syncthreads();
  int n = blockIdx.x * blockDim.x + t;
  if (n < Ntok) {
    const float* xr = x + (size_t)n * DD;
    float lg[8];
    for (int e = 0; e < 8; ++e) {
      const float* gw = gate_w + (size_t)e * DD;
      float s0 = 0.f, s1 = 0.f, s2 = 0.f, s3 = 0.f;
      for (int d = 0; d < DD; d += 4) {
        s0 += xr[d] * gw[d];         s1 += xr[d + 1] * gw[d + 1];
        s2 += xr[d + 2] * gw[d + 2]; s3 += xr[d + 3] * gw[d + 3];
      }
      lg[e] = (s0 + s1) + (s2 + s3);
    }
    int i1 = 0; float v1 = lg[0];
    for (int e = 1; e < 8; ++e) if (lg[e] > v1) { v1 = lg[e]; i1 = e; }
    int i2 = (i1 == 0) ? 1 : 0; float v2 = lg[i2];
    for (int e = 0; e < 8; ++e) if (e != i1 && lg[e] > v2) { v2 = lg[e]; i2 = e; }
    float e2 = expf(v2 - v1);
    float p0 = 1.0f / (1.0f + e2), p1 = e2 / (1.0f + e2);
    te[n * 2] = i1;  te[n * 2 + 1] = i2;
    prio[n * 2] = v1; prio[n * 2 + 1] = v2;
    tpq[n * 2] = p0;  tpq[n * 2 + 1] = p1;
    // full router softmax + z-loss
    float m = lg[0];
    for (int e = 1; e < 8; ++e) m = fmaxf(m, lg[e]);
    float se = 0.f, ex[8];
    for (int e = 0; e < 8; ++e) { ex[e] = expf(lg[e] - m); se += ex[e]; }
    float inv = 1.0f / se;
    for (int e = 0; e < 8; ++e) atomicAdd(&sAcc[e], ex[e] * inv);
    atomicAdd(&sAcc[8 + i1], p0);
    atomicAdd(&sAcc[8 + i2], p1);
    float z = m + logf(se);
    atomicAdd(&sAcc[16], z * z);
  }
  __syncthreads();
  if (t < 18) atomicAdd(&accum[t], sAcc[t]);
}

// ---------------- exact priority ranking (lexsort semantics) ----------------
__global__ __launch_bounds__(256)
void k_rank(const int* te, const float* prio, int* slotA, float* accum,
            int NKtot, int Cap) {
  __shared__ int   ste[LDS_CHUNK];
  __shared__ float spr[LDS_CHUNK];
  int i = blockIdx.x * blockDim.x + threadIdx.x;
  int e = te[i]; float pv = prio[i];
  int rank = 0;
  for (int base = 0; base < NKtot; base += LDS_CHUNK) {
    __syncthreads();
    for (int j = threadIdx.x; j < LDS_CHUNK; j += blockDim.x) {
      ste[j] = te[base + j]; spr[j] = prio[base + j];
    }
    __syncthreads();
    for (int j = 0; j < LDS_CHUNK; ++j) {
      int jj = base + j;
      bool higher = (spr[j] > pv) || (spr[j] == pv && jj < i);
      rank += (ste[j] == e && higher) ? 1 : 0;
    }
  }
  int s = (rank < Cap) ? (e * Cap + rank) : -1;
  slotA[i] = s;
  if (s >= 0) atomicAdd(&accum[17], 1.0f);
}

// ---------------- dispatch scatter: x row -> expert buffer slot ----------------
__global__ __launch_bounds__(256)
void k_scatter(const float* x, const int* slotA, float* buf, int NKtot) {
  int w = blockIdx.x * 8 + (threadIdx.x >> 5);
  int lane = threadIdx.x & 31;
  if (w >= NKtot) return;
  int s = slotA[w];
  if (s < 0) return;
  const float4* src = (const float4*)(x + (size_t)(w >> 1) * DD);
  float4* dst = (float4*)(buf + (size_t)s * DD);
  for (int i = lane; i < DD / 4; i += 32) dst[i] = src[i];
}

__global__ void k_weights(const float* tpq, const int* slotA, float* wts, int Ntok) {
  int n = blockIdx.x * blockDim.x + threadIdx.x;
  if (n >= Ntok) return;
  float k0 = slotA[n * 2]     >= 0 ? 1.f : 0.f;
  float k1 = slotA[n * 2 + 1] >= 0 ? 1.f : 0.f;
  float t0 = tpq[n * 2] * k0, t1 = tpq[n * 2 + 1] * k1;
  float dn = t0 + t1;
  float inv = dn > 0.f ? 1.0f / dn : 0.0f;
  wts[n * 2] = t0 * inv; wts[n * 2 + 1] = t1 * inv;
}

// ---------------- generic WMMA GEMM: C = act(alpha*A@B^T + bias + Res) ----------
// block = 8 waves as 4(M) x 2(N); wave tile = 16(M) x 64(N): 4 accumulators,
// A fragment reused 4x. K-loop unrolled 2x as a ping-pong software pipeline:
// two disjoint fragment register sets, each loaded while the other feeds the
// 4 WMMAs (8 v_wmma per loop body, no rotation movs). Requires Kd % 64 == 0
// (true for all launches: 768, 2048). Last preload uses a scalar-uniform
// clamped index -> branch-free, EXEC all-ones at every WMMA.
__global__ __launch_bounds__(256)
void k_gemm(const float* A, int lda, const _Float16* B, int ldb,
            float* Cc, int ldc, const float* bias,
            const float* Res, int ldr, int Kd, float alpha, int act) {
  int wave = threadIdx.x >> 5, lane = threadIdx.x & 31;
  int tileM  = blockIdx.y * 64  + (wave >> 1) * 16;
  int tileN0 = blockIdx.x * 128 + (wave & 1) * 64;
  int half = lane >> 4, l15 = lane & 15;
  const float* Arow = A + (size_t)(tileM + l15) * lda;
  const _Float16* B0 = B + (size_t)(tileN0 + l15) * ldb;
  const _Float16* B1 = B0 + (size_t)16 * ldb;
  const _Float16* B2 = B0 + (size_t)32 * ldb;
  const _Float16* B3 = B0 + (size_t)48 * ldb;
  v8f acc0 = {}, acc1 = {}, acc2 = {}, acc3 = {};
  int kh = half * 8;
  // stage 0 preload (kt = 0)
  v16h a0  = load_fragA(Arow, kh);
  v16h p00 = load_fragB(B0, kh);
  v16h p01 = load_fragB(B1, kh);
  v16h p02 = load_fragB(B2, kh);
  v16h p03 = load_fragB(B3, kh);
  for (int kt = 0; kt < Kd; kt += 64) {
    // stage 1 load (kt+32, always in-bounds since Kd % 64 == 0)
    int k1 = kt + 32 + kh;
    if (kt + 128 < Kd) __builtin_prefetch(Arow + kt + 128, 0, 3); // global_prefetch_b8
    v16h a1  = load_fragA(Arow, k1);
    v16h p10 = load_fragB(B0, k1);
    v16h p11 = load_fragB(B1, k1);
    v16h p12 = load_fragB(B2, k1);
    v16h p13 = load_fragB(B3, k1);
    // consume stage 0
    acc0 = wmma_f16(a0, p00, acc0);
    acc1 = wmma_f16(a0, p01, acc1);
    acc2 = wmma_f16(a0, p02, acc2);
    acc3 = wmma_f16(a0, p03, acc3);
    // stage 0 load for next iteration (clamped on final iteration)
    int k2 = ((kt + 64 < Kd) ? (kt + 64) : 0) + kh;   // uniform clamp, no branch
    a0  = load_fragA(Arow, k2);
    p00 = load_fragB(B0, k2);
    p01 = load_fragB(B1, k2);
    p02 = load_fragB(B2, k2);
    p03 = load_fragB(B3, k2);
    // consume stage 1
    acc0 = wmma_f16(a1, p10, acc0);
    acc1 = wmma_f16(a1, p11, acc1);
    acc2 = wmma_f16(a1, p12, acc2);
    acc3 = wmma_f16(a1, p13, acc3);
  }
  // C/D layout: VGPR r: lanes0-15 -> (M=tileM+r, N=lane), lanes16-31 -> (M=+8+r)
  int cm0 = tileM + half * 8;
  v8f accs[4] = {acc0, acc1, acc2, acc3};
#pragma unroll
  for (int j = 0; j < 4; ++j) {
    int cn = tileN0 + j * 16 + l15;
    float bv = bias ? bias[cn] : 0.0f;
#pragma unroll
    for (int r = 0; r < 8; ++r) {
      float v = alpha * accs[j][r] + bv;
      if (Res) v += Res[(size_t)(cm0 + r) * ldr + cn];
      if (act == 1) v = 0.5f * v * (1.0f + erff(v * 0.70710678118f)); // exact GELU
      Cc[(size_t)(cm0 + r) * ldc + cn] = v;
    }
  }
}

// ---------------- fused expert FFN stage 1: h = silu(2A@Wg^T) * (2A@Wu^T) -------
// wave tile = 16(M) x 32(N) with gate+up halves; same 2x ping-pong pipeline,
// 8 v_wmma per loop body. Kd = 768 (multiple of 64).
__global__ __launch_bounds__(256)
void k_ffn1(const float* A, const _Float16* W, float* Hc, int Hn, int Kd,
            float alpha) {
  int wave = threadIdx.x >> 5, lane = threadIdx.x & 31;
  int tileM  = blockIdx.y * 64 + (wave >> 1) * 16;
  int tileN0 = blockIdx.x * 64 + (wave & 1) * 32;
  int half = lane >> 4, l15 = lane & 15;
  const float* Arow = A + (size_t)(tileM + l15) * Kd;
  const _Float16* Bg0 = W + (size_t)(tileN0 + l15) * Kd;        // gate rows [0,H)
  const _Float16* Bg1 = Bg0 + (size_t)16 * Kd;
  const _Float16* Bu0 = W + (size_t)(Hn + tileN0 + l15) * Kd;   // up rows [H,2H)
  const _Float16* Bu1 = Bu0 + (size_t)16 * Kd;
  v8f accg0 = {}, accg1 = {}, accu0 = {}, accu1 = {};
  int kh = half * 8;
  v16h a0  = load_fragA(Arow, kh);
  v16h g00 = load_fragB(Bg0, kh);
  v16h g01 = load_fragB(Bg1, kh);
  v16h u00 = load_fragB(Bu0, kh);
  v16h u01 = load_fragB(Bu1, kh);
  for (int kt = 0; kt < Kd; kt += 64) {
    int k1 = kt + 32 + kh;
    if (kt + 128 < Kd) __builtin_prefetch(Arow + kt + 128, 0, 3);
    v16h a1  = load_fragA(Arow, k1);
    v16h g10 = load_fragB(Bg0, k1);
    v16h g11 = load_fragB(Bg1, k1);
    v16h u10 = load_fragB(Bu0, k1);
    v16h u11 = load_fragB(Bu1, k1);
    accg0 = wmma_f16(a0, g00, accg0);
    accg1 = wmma_f16(a0, g01, accg1);
    accu0 = wmma_f16(a0, u00, accu0);
    accu1 = wmma_f16(a0, u01, accu1);
    int k2 = ((kt + 64 < Kd) ? (kt + 64) : 0) + kh;
    a0  = load_fragA(Arow, k2);
    g00 = load_fragB(Bg0, k2);
    g01 = load_fragB(Bg1, k2);
    u00 = load_fragB(Bu0, k2);
    u01 = load_fragB(Bu1, k2);
    accg0 = wmma_f16(a1, g10, accg0);
    accg1 = wmma_f16(a1, g11, accg1);
    accu0 = wmma_f16(a1, u10, accu0);
    accu1 = wmma_f16(a1, u11, accu1);
  }
  int cm0 = tileM + half * 8;
  v8f ag[2] = {accg0, accg1}, au[2] = {accu0, accu1};
#pragma unroll
  for (int j = 0; j < 2; ++j) {
    int cn = tileN0 + j * 16 + l15;
#pragma unroll
    for (int r = 0; r < 8; ++r) {
      float g = alpha * ag[j][r], u = alpha * au[j][r];
      Hc[(size_t)(cm0 + r) * Hn + cn] = (g / (1.0f + expf(-g))) * u;  // silu(g)*u
    }
  }
}

// ---------------- token group assembly: [expert_out_k0, expert_out_k1, mediator] --
__global__ __launch_bounds__(256)
void k_init_tokens(const float* buf, const int* slotA, const float* mediator,
                   float* tokens, int n0) {
  int w = blockIdx.x * 8 + (threadIdx.x >> 5);
  int lane = threadIdx.x & 31;
  int nl = w / 3, l = w - nl * 3;
  float4* dst = (float4*)(tokens + (size_t)w * DD);
  const float4* src;
  if (l == 2) {
    src = (const float4*)mediator;
  } else {
    int s = slotA[(n0 + nl) * 2 + l];
    if (s < 0) {
      float4 z = {0.f, 0.f, 0.f, 0.f};
      for (int i = lane; i < DD / 4; i += 32) dst[i] = z;
      return;
    }
    src = (const float4*)(buf + (size_t)s * DD);
  }
  for (int i = lane; i < DD / 4; i += 32) dst[i] = src[i];
}

// ---------------- 3-token multi-head attention, wave per (token,head) ----------
__global__ __launch_bounds__(256)
void k_attn(const float* qkv, const int* slotA, float* o, int n0) {
  int w = blockIdx.x * 8 + (threadIdx.x >> 5);
  int lane = threadIdx.x & 31;
  int nl = w >> 2, hh = w & 3;
  const float* base = qkv + (size_t)nl * 3 * TD + hh * DHD;
  float q[3][6], kk[3][6], vv[3][6];
#pragma unroll
  for (int l = 0; l < 3; ++l)
#pragma unroll
    for (int t = 0; t < 6; ++t) {
      size_t idx = (size_t)l * TD + lane + 32 * t;
      q[l][t]  = base[idx];
      kk[l][t] = base[idx + DD];
      vv[l][t] = base[idx + 2 * DD];
    }
  const float scale = 0.072168783649f;  // 1/sqrt(192)
  int n = n0 + nl;
  float b0 = (slotA[n * 2]     >= 0) ? 0.0f : -1e9f;
  float b1 = (slotA[n * 2 + 1] >= 0) ? 0.0f : -1e9f;
  float a[3][3];
#pragma unroll
  for (int i = 0; i < 3; ++i) {
    float s[3];
#pragma unroll
    for (int j = 0; j < 3; ++j) {
      float ps = 0.f;
#pragma unroll
      for (int t = 0; t < 6; ++t) ps += q[i][t] * kk[j][t];
      s[j] = wave_sum(ps) * scale;
    }
    s[0] += b0; s[1] += b1;
    float mx = fmaxf(fmaxf(s[0], s[1]), s[2]);
    float e0 = expf(s[0] - mx), e1 = expf(s[1] - mx), e2 = expf(s[2] - mx);
    float inv = 1.0f / (e0 + e1 + e2);
    a[i][0] = e0 * inv; a[i][1] = e1 * inv; a[i][2] = e2 * inv;
  }
#pragma unroll
  for (int i = 0; i < 3; ++i)
#pragma unroll
    for (int t = 0; t < 6; ++t) {
      float ov = a[i][0] * vv[0][t] + a[i][1] * vv[1][t] + a[i][2] * vv[2][t];
      o[(size_t)(nl * 3 + i) * DD + hh * DHD + lane + 32 * t] = ov;
    }
}

// ---------------- RMS norm, wave per row ----------------
__global__ __launch_bounds__(256)
void k_rms(const float* in, const float* w, float* out) {
  int row = blockIdx.x * 8 + (threadIdx.x >> 5);
  int lane = threadIdx.x & 31;
  const float* r = in + (size_t)row * DD;
  float ss = 0.f;
#pragma unroll
  for (int t = 0; t < 24; ++t) { float v = r[lane + 32 * t]; ss += v * v; }
  ss = wave_sum(ss);
  float sc = rsqrtf(ss * (1.0f / DD) + 1e-6f);
  float* og = out + (size_t)row * DD;
#pragma unroll
  for (int t = 0; t < 24; ++t) { int d = lane + 32 * t; og[d] = r[d] * sc * w[d]; }
}

// ---------------- fuse: g*med + (1-g)*sum_k w_k*ref_k ----------------
__global__ __launch_bounds__(256)
void k_fuse(const float* tokens, const float* wts, const float* fgw,
            const float* fgb, float* out, int n0) {
  int nl = blockIdx.x * 8 + (threadIdx.x >> 5);
  int lane = threadIdx.x & 31;
  const float* r0  = tokens + (size_t)(nl * 3) * DD;
  const float* r1  = tokens + (size_t)(nl * 3 + 1) * DD;
  const float* med = tokens + (size_t)(nl * 3 + 2) * DD;
  float dp = 0.f;
#pragma unroll
  for (int t = 0; t < 24; ++t) { int d = lane + 32 * t; dp += med[d] * fgw[d]; }
  dp = wave_sum(dp) + fgb[0];
  float g = 1.0f / (1.0f + expf(-dp));
  int n = n0 + nl;
  float w0 = wts[n * 2], w1 = wts[n * 2 + 1];
  float* og = out + (size_t)nl * DD;
#pragma unroll
  for (int t = 0; t < 24; ++t) {
    int d = lane + 32 * t;
    og[d] = g * med[d] + (1.0f - g) * (w0 * r0[d] + w1 * r1[d]);
  }
}

// ---------------- aux scalar + top-k index output ----------------
__global__ void k_final(const float* accum, const int* te, float* outp,
                        int Ntok, int NKtot) {
  int i = blockIdx.x * blockDim.x + threadIdx.x;
  if (i == 0) {
    float bal = 0.f;
    for (int e = 0; e < 8; ++e)
      bal += (accum[e] / Ntok) * (accum[8 + e] / Ntok);
    bal *= 8.0f;
    float z2m  = accum[16] / Ntok;
    float drop = 1.0f - accum[17] / NKtot;
    outp[(size_t)Ntok * DD] = 0.01f * bal + 0.001f * z2m + 0.001f * drop;
  }
  if (i < NKtot) {
    int* ip = (int*)(outp + (size_t)Ntok * DD + 1);
    ip[i] = te[i];
  }
}

// ---------------- host side ----------------
static void launch_gemm(hipStream_t s, const float* A, int lda, const _Float16* B,
                        int ldb, float* Cc, int ldc, const float* bias,
                        const float* Res, int ldr, int M, int Nn, int Kd,
                        float alpha, int act) {
  dim3 g(Nn / 128, M / 64);
  k_gemm<<<g, 256, 0, s>>>(A, lda, B, ldb, Cc, ldc, bias, Res, ldr, Kd, alpha, act);
}

extern "C" void kernel_launch(void* const* d_in, const int* in_sizes, int n_in,
                              void* d_out, int out_size, void* d_ws, size_t ws_size,
                              hipStream_t stream) {
  (void)n_in; (void)out_size; (void)ws_size;
  const float* x         = (const float*)d_in[0];
  const float* gate_w    = (const float*)d_in[1];
  const float* w13       = (const float*)d_in[2];
  const float* w2        = (const float*)d_in[3];
  const float* in_proj_w = (const float*)d_in[4];
  const float* in_proj_b = (const float*)d_in[5];
  const float* out_w     = (const float*)d_in[6];
  const float* out_b     = (const float*)d_in[7];
  const float* norm1_w   = (const float*)d_in[8];
  const float* norm2_w   = (const float*)d_in[9];
  const float* ffn_w1    = (const float*)d_in[10];
  const float* ffn_w2    = (const float*)d_in[11];
  const float* mediator  = (const float*)d_in[12];
  const float* fgw       = (const float*)d_in[13];
  const float* fgb       = (const float*)d_in[14];
  const float* o_proj_w  = (const float*)d_in[15];
  float* outp = (float*)d_out;

  const int N  = in_sizes[0] / DD;   // 8192 tokens
  const int NK = N * KK;             // 16384 entries
  int capMin = (NK / EE) + (NK / EE) / 4;       // ceil(NK/E * 1.25)
  int Cap = 1; while (Cap < capMin) Cap <<= 1;  // 4096
  const int nChunks = N / CHUNK;
  const int CL3 = CHUNK * 3;

  // workspace carve-out
  char* p = (char*)d_ws;
  auto take = [&](size_t bytes) -> char* {
    char* r = p; p += (bytes + 255) & ~(size_t)255; return r;
  };
  int*   te    = (int*)  take((size_t)NK * 4);
  float* prio  = (float*)take((size_t)NK * 4);
  float* tpq   = (float*)take((size_t)NK * 4);
  int*   slotA = (int*)  take((size_t)NK * 4);
  float* accum = (float*)take(32 * 4);
  float* wts   = (float*)take((size_t)NK * 4);
  float* buf   = (float*)take((size_t)EE * Cap * DD * 4);  // dispatch + y (in place)
  float* hbuf  = (float*)take((size_t)Cap * HFF * 4);      // per-expert hidden
  float* tokens= (float*)take((size_t)CL3 * DD * 4);
  float* tkb   = (float*)take((size_t)CL3 * DD * 4);
  float* tkc   = (float*)take((size_t)CL3 * DD * 4);
  float* qkvb  = (float*)take((size_t)CL3 * TD * 4);
  // f16 weight shadows (converted once per call)
  _Float16* w13h = (_Float16*)take((size_t)EE * 2 * HFF * DD * 2);
  _Float16* w2h  = (_Float16*)take((size_t)EE * DD * HFF * 2);
  _Float16* ipwh = (_Float16*)take((size_t)TD * DD * 2);
  _Float16* owh  = (_Float16*)take((size_t)DD * DD * 2);
  _Float16* f1h  = (_Float16*)take((size_t)DD * DD * 2);
  _Float16* f2h  = (_Float16*)take((size_t)DD * DD * 2);
  _Float16* oph  = (_Float16*)take((size_t)DD * DD * 2);

  hipMemsetAsync(accum, 0, 32 * 4, stream);
  hipMemsetAsync(buf, 0, (size_t)EE * Cap * DD * 4, stream);

  // weight f32 -> f16 (element counts all divisible by 1024)
  auto cvt = [&](const float* s, _Float16* d, size_t cnt) {
    k_cvt16<<<(unsigned)(cnt / 1024), 256, 0, stream>>>(s, d);
  };
  cvt(w13,       w13h, (size_t)EE * 2 * HFF * DD);
  cvt(w2,        w2h,  (size_t)EE * DD * HFF);
  cvt(in_proj_w, ipwh, (size_t)TD * DD);
  cvt(out_w,     owh,  (size_t)DD * DD);
  cvt(ffn_w1,    f1h,  (size_t)DD * DD);
  cvt(ffn_w2,    f2h,  (size_t)DD * DD);
  cvt(o_proj_w,  oph,  (size_t)DD * DD);

  // gating / ranking / dispatch
  k_gate<<<N / 256, 256, 0, stream>>>(x, gate_w, te, prio, tpq, accum, N);
  k_rank<<<NK / 256, 256, 0, stream>>>(te, prio, slotA, accum, NK, Cap);
  k_scatter<<<NK / 8, 256, 0, stream>>>(x, slotA, buf, NK);
  k_weights<<<N / 256, 256, 0, stream>>>(tpq, slotA, wts, N);

  // expert FFN: h = silu(2*buf@Wg^T)*(2*buf@Wu^T); y = buf + h@W2^T (in place)
  for (int e = 0; e < EE; ++e) {
    float* bufe = buf + (size_t)e * Cap * DD;
    k_ffn1<<<dim3(HFF / 64, Cap / 64), 256, 0, stream>>>(
        bufe, w13h + (size_t)e * 2 * HFF * DD, hbuf, HFF, DD, 2.0f);
    launch_gemm(stream, hbuf, HFF, w2h + (size_t)e * DD * HFF, HFF,
                bufe, DD, nullptr, bufe, DD, Cap, DD, HFF, 1.0f, 0);
  }

  // collaborative refinement + fuse, chunked over tokens
  for (int c = 0; c < nChunks; ++c) {
    int n0 = c * CHUNK;
    k_init_tokens<<<CL3 / 8, 256, 0, stream>>>(buf, slotA, mediator, tokens, n0);
    for (int step = 0; step < 2; ++step) {
      launch_gemm(stream, tokens, DD, ipwh, DD, qkvb, TD,
                  in_proj_b, nullptr, 0, CL3, TD, DD, 1.0f, 0);
      k_attn<<<(CHUNK * NHEADS) / 8, 256, 0, stream>>>(qkvb, slotA, tkb, n0);
      launch_gemm(stream, tkb, DD, owh, DD, tkc, DD,
                  out_b, tokens, DD, CL3, DD, DD, 1.0f, 0);
      k_rms<<<CL3 / 8, 256, 0, stream>>>(tkc, norm1_w, tokens);
      k_rms<<<CL3 / 8, 256, 0, stream>>>(tokens, norm2_w, tkb);
      launch_gemm(stream, tkb, DD, f1h, DD, tkc, DD,
                  nullptr, nullptr, 0, CL3, DD, DD, 1.0f, 1 /*gelu*/);
      launch_gemm(stream, tkc, DD, f2h, DD, tokens, DD,
                  nullptr, tokens, DD, CL3, DD, DD, 1.0f, 0);
    }
    k_fuse<<<CHUNK / 8, 256, 0, stream>>>(tokens, wts, fgw, fgb, tkb, n0);
    launch_gemm(stream, tkb, DD, oph, DD, outp + (size_t)n0 * DD, DD,
                nullptr, nullptr, 0, CHUNK, DD, DD, 1.0f, 0);
  }

  k_final<<<(NK + 255) / 256, 256, 0, stream>>>(accum, te, outp, N, NK);
}
